// OCSVMLayer_62551903699266
// MI455X (gfx1250) — compile-verified, matrix operand order checked by reading, not run on hardware
//
#include <hip/hip_runtime.h>
#include <hip/hip_bf16.h>
#include <stdint.h>

// ---------------------------------------------------------------------------
// OCSVM RBF decision function, fused:  score[b] = sum_s c[s]*exp(-g*d2) - rho
// d2 = |x|^2 + |s|^2 - 2 * (X @ SV^T)   -- GEMM done in bf16 WMMA, f32 accum.
// Double-buffered LDS pipeline: global loads for tile i+1 overlap WMMA on i.
// ---------------------------------------------------------------------------

typedef __attribute__((ext_vector_type(16))) __bf16        v16bf;
typedef __attribute__((ext_vector_type(8)))  float         v8f;
typedef __attribute__((ext_vector_type(4)))  __bf16        v4bf;
typedef __attribute__((ext_vector_type(4)))  float         f32x4;
typedef __attribute__((ext_vector_type(4)))  unsigned int  u32x4;

#define M_TILE  64     // rows of inputs per workgroup
#define N_TILE  128    // support vectors per S-chunk
#define K_TILE  64     // bf16 K per LDS stage (2 WMMA k-steps)
#define KSTRIDE 72     // padded LDS row stride in bf16 (144B rows, 16B-aligned)

union Frag { v16bf v; u32x4 u[2]; };

// ---------------- kernel 1: fp32 -> bf16 + row sum of squares --------------
__global__ void __launch_bounds__(128)
convert_rowsq_kernel(const float* __restrict__ src, __bf16* __restrict__ dst,
                     float* __restrict__ rowsq, int cols) {
  const int row = blockIdx.x;
  const int t   = threadIdx.x;
  const float*  srow = src + (size_t)row * cols;
  __bf16*       drow = dst + (size_t)row * cols;

  float ss = 0.f;
  for (int c = t * 4; c < cols; c += 128 * 4) {
    f32x4 v = *(const f32x4*)(srow + c);
    ss += v.x * v.x + v.y * v.y + v.z * v.z + v.w * v.w;
    v4bf h;
    h.x = (__bf16)v.x; h.y = (__bf16)v.y;
    h.z = (__bf16)v.z; h.w = (__bf16)v.w;
    *(v4bf*)(drow + c) = h;
  }

  __shared__ float red[128];
  red[t] = ss;
  __syncthreads();
  for (int s = 64; s > 0; s >>= 1) {
    if (t < s) red[t] += red[t + s];
    __syncthreads();
  }
  if (t == 0) rowsq[row] = red[0];
}

// ---------------- kernel 2: fused WMMA GEMM + RBF epilogue -----------------
__global__ void __launch_bounds__(256)
ocsvm_fused_kernel(const __bf16* __restrict__ xbf, const __bf16* __restrict__ sbf,
                   const float* __restrict__ xsq, const float* __restrict__ ssq,
                   const float* __restrict__ coeff, const float* __restrict__ rho,
                   const float* __restrict__ gammap, float* __restrict__ out,
                   int S, int F) {
  __shared__ __align__(16) __bf16 As[2 * M_TILE * KSTRIDE];   // 18,432 B
  __shared__ __align__(16) __bf16 Bs[2 * N_TILE * KSTRIDE];   // 36,864 B
  __shared__ float red[M_TILE][32];                           //  8,192 B

  const int tid    = threadIdx.x;
  const int lane   = tid & 31;
  const int wave   = tid >> 5;
  const int wave_m = wave & 3;          // 0..3  -> 16-row slice inside M_TILE
  const int wave_n = wave >> 2;         // 0..1  -> 64-col slice inside N_TILE
  const int half   = (lane >> 4) & 1;   // wave32 half (C-layout: M = r + 8*half)
  const int l16    = lane & 15;
  const int m_block = blockIdx.x * M_TILE;

  const float gamma = gammap[0];

  // per-lane |x|^2 values matching the WMMA C-matrix row mapping
  float xsq_r[8];
#pragma unroll
  for (int r = 0; r < 8; ++r)
    xsq_r[r] = xsq[m_block + wave_m * 16 + half * 8 + r];

  float runsum[8];
#pragma unroll
  for (int r = 0; r < 8; ++r) runsum[r] = 0.f;

  // tile-staging geometry: 16B chunks, 4 per 64-wide row
  const int rowA = tid >> 2;
  const int c8   = (tid & 3) * 8;
  const __bf16* xrow = xbf + (size_t)(m_block + rowA) * F + c8;

  const v8f zero = {};
  v8f acc[4];

  const int n_chunks = S / N_TILE;
  for (int sc = 0; sc < n_chunks; ++sc) {
#pragma unroll
    for (int t = 0; t < 4; ++t) acc[t] = zero;

    const __bf16* brow = sbf + (size_t)(sc * N_TILE + rowA) * F + c8;

    // pipeline prologue: fetch K-tile 0 into registers
    u32x4 ga = *(const u32x4*)(xrow);
    u32x4 g0 = *(const u32x4*)(brow);
    u32x4 g1 = *(const u32x4*)(brow + (size_t)64 * F);

    for (int kb = 0; kb < F; kb += K_TILE) {
      const int buf = (kb >> 6) & 1;
      __bf16* Ab = As + buf * (M_TILE * KSTRIDE);
      __bf16* Bb = Bs + buf * (N_TILE * KSTRIDE);

      // commit staged registers to LDS buffer `buf`
      *(u32x4*)(Ab + rowA * KSTRIDE + c8)        = ga;
      *(u32x4*)(Bb + rowA * KSTRIDE + c8)        = g0;
      *(u32x4*)(Bb + (rowA + 64) * KSTRIDE + c8) = g1;
      __syncthreads();

      // immediately issue global loads for the NEXT K-tile (overlaps WMMAs)
      if (kb + K_TILE < F) {
        ga = *(const u32x4*)(xrow + kb + K_TILE);
        g0 = *(const u32x4*)(brow + kb + K_TILE);
        g1 = *(const u32x4*)(brow + (size_t)64 * F + kb + K_TILE);
      }

#pragma unroll
      for (int kk = 0; kk < K_TILE; kk += 32) {
        // A fragment: lane holds row M=l16, K = k0..k0+7 and k0+16..k0+23
        const int k0 = kk + half * 8;
        Frag af;
        const __bf16* ap = Ab + (wave_m * 16 + l16) * KSTRIDE + k0;
        af.u[0] = *(const u32x4*)ap;
        af.u[1] = *(const u32x4*)(ap + 16);
        // load ALL B fragments first so the 4 WMMAs issue back-to-back
        Frag bfr[4];
#pragma unroll
        for (int t = 0; t < 4; ++t) {
          const __bf16* bp = Bb + (wave_n * 64 + t * 16 + l16) * KSTRIDE + k0;
          bfr[t].u[0] = *(const u32x4*)bp;
          bfr[t].u[1] = *(const u32x4*)(bp + 16);
        }
#pragma unroll
        for (int t = 0; t < 4; ++t)
          acc[t] = __builtin_amdgcn_wmma_f32_16x16x32_bf16(
              false, af.v, false, bfr[t].v, (short)0, acc[t], false, false);
      }
    }

    // epilogue for this 128-wide S chunk: k = exp(-g*d2), weighted accumulate
#pragma unroll
    for (int t = 0; t < 4; ++t) {
      const int n  = sc * N_TILE + wave_n * 64 + t * 16 + l16;
      const float sq = ssq[n];
      const float co = coeff[n];
#pragma unroll
      for (int r = 0; r < 8; ++r) {
        float d2 = xsq_r[r] + sq - 2.0f * acc[t][r];
        runsum[r] += co * __expf(-gamma * d2);
      }
    }
    __syncthreads();  // keep double-buffer safe for any F/K_TILE parity
  }

  // deterministic cross-lane / cross-wave_n reduction through LDS
#pragma unroll
  for (int r = 0; r < 8; ++r)
    red[wave_m * 16 + half * 8 + r][wave_n * 16 + l16] = runsum[r];
  __syncthreads();
  if (tid < M_TILE) {
    float s = 0.f;
#pragma unroll 8
    for (int j = 0; j < 32; ++j) s += red[tid][j];
    out[m_block + tid] = s - rho[0];
  }
}

// ---------------------------------------------------------------------------
extern "C" void kernel_launch(void* const* d_in, const int* in_sizes, int n_in,
                              void* d_out, int out_size, void* d_ws, size_t ws_size,
                              hipStream_t stream) {
  const float* inputs = (const float*)d_in[0];   // [B, F] fp32
  const float* sv     = (const float*)d_in[1];   // [S, F] fp32
  const float* coeff  = (const float*)d_in[2];   // [S]    fp32
  const float* rho    = (const float*)d_in[3];   // [1]
  const float* gamma  = (const float*)d_in[4];   // [1]
  float* out = (float*)d_out;

  const int B = out_size;          // 16384
  const int S = in_sizes[2];       // 8192
  const int F = in_sizes[0] / B;   // 512

  // workspace layout: bf16 inputs | bf16 SVs | x_sq | s_sq  (~25.3 MB)
  __bf16* xbf = (__bf16*)d_ws;
  __bf16* sbf = xbf + (size_t)B * F;
  float*  xsq = (float*)(sbf + (size_t)S * F);
  float*  ssq = xsq + B;
  (void)ws_size; (void)n_in;

  convert_rowsq_kernel<<<B, 128, 0, stream>>>(inputs, xbf, xsq, F);
  convert_rowsq_kernel<<<S, 128, 0, stream>>>(sv, sbf, ssq, F);
  ocsvm_fused_kernel<<<B / M_TILE, 256, 0, stream>>>(
      xbf, sbf, xsq, ssq, coeff, rho, gamma, out, S, F);
}
